// CondSpline1D_6133213298945
// MI455X (gfx1250) — matrix operand order repaired and checked by reading, not compiled, not run on hardware
//
#include <hip/hip_runtime.h>
#include <hip/hip_bf16.h>

// ---------------- problem constants ----------------
#define K_BINS 64
#define HID    32
#define PDIM   255        // 4*K-1
#define BOUND  3.0f
#define MINW   0.001f
#define MINH   0.001f
#define MIND   0.001f

// ---------------- tiling constants -----------------
#define BLOCK  256        // threads per block (8 wave32)
#define ELEMS  256        // elements per block (32 per wave)
#define NT3    16         // layer-3 N tiles (255 cols padded to 256)
#define PSTR   260        // params LDS stride in elements (pad, %4==0 for b128)
#define HSTR   40         // h LDS stride in halves (80B rows, 16B aligned)

typedef __attribute__((ext_vector_type(16))) _Float16 v16h;
typedef __attribute__((ext_vector_type(8)))  float    v8f;
typedef __attribute__((ext_vector_type(4)))  float    f32x4;
typedef __attribute__((ext_vector_type(4)))  unsigned int u32x4;
typedef __attribute__((ext_vector_type(8)))  int      i32x8;
typedef __attribute__((ext_vector_type(4)))  int      i32x4;

union HFrag { v16h v; u32x4 q[2]; };

__device__ __forceinline__ float softplus_f(float t) {
    // log1p(exp(t)) stable: max(t,0) + log(1+exp(-|t|))
    return fmaxf(t, 0.f) + __logf(1.f + __expf(-fabsf(t)));
}

// ---- Tensor Data Mover: stage W3 (32 x 255 f32, row-major) into LDS ----
// D# per CDNA5 ISA ch.8: group0 {count=1, lds_addr, global_addr[56:0], type=2},
// group1 {data_size=4B, tensor_dim0=255, tensor_dim1=32, tile_dim0=255,
//         tile_dim1=32, tensor_dim0_stride=255}; groups 2/3 zero (2-D tensor).
__device__ __forceinline__ void tdm_load_w3(const float* gsrc, unsigned lds_off) {
    const unsigned long long ga = (unsigned long long)(uintptr_t)gsrc;
    u32x4 g0;
    g0[0] = 1u;                                             // count=1 (valid user D#)
    g0[1] = lds_off;                                        // lds_addr (bytes)
    g0[2] = (unsigned)(ga & 0xFFFFFFFFu);                   // global_addr[31:0]
    g0[3] = (unsigned)((ga >> 32) & 0x01FFFFFFu) | (2u << 30); // addr[56:32] | type=2
    i32x8 g1;
    g1[0] = 0x20000;                                        // data_size=2 (4 bytes)
    g1[1] = (int)(((unsigned)PDIM & 0xFFFFu) << 16);        // tensor_dim0[15:0] @bit48
    g1[2] = (int)(((unsigned)PDIM >> 16) | ((unsigned)HID << 16)); // dim0 hi | dim1 lo
    g1[3] = (int)(((unsigned)HID >> 16) | ((unsigned)PDIM << 16)); // dim1 hi | tile_dim0
    g1[4] = HID;                                            // tile_dim1 (tile_dim2=0)
    g1[5] = PDIM;                                           // tensor_dim0_stride lo32
    g1[6] = 0;
    g1[7] = 0;
    const i32x4 gz4 = {0, 0, 0, 0};
    const i32x8 gz8 = {0, 0, 0, 0, 0, 0, 0, 0};
    // clang-23 / therock-10.0 toolchain: 6-arg form
    __builtin_amdgcn_tensor_load_to_lds(g0, g1, gz4, gz4, gz8, 0);
}

__global__ __launch_bounds__(BLOCK, 1)
void cond_spline_kernel(const float* __restrict__ x,  const float* __restrict__ condx,
                        const float* __restrict__ W1, const float* __restrict__ b1,
                        const float* __restrict__ W2, const float* __restrict__ b2,
                        const float* __restrict__ W3, const float* __restrict__ b3,
                        float* __restrict__ out, int n)
{
    // ---- LDS ----
    __shared__ float    sP [NT3 * 16 * PSTR];   // params^T: [col][elem]; also TDM staging
    __shared__ _Float16 sH [ELEMS * HSTR];      // h1/h2 rows (row=elem, col=K)
    __shared__ _Float16 sW3[NT3 * 32 * 16];     // W3 B-fragments, per-lane contiguous
    __shared__ _Float16 sW2[2   * 32 * 16];     // W2 B-fragments
    __shared__ float    sB3[256];
    __shared__ float    sB2[HID];
    __shared__ float    sW1v[HID];
    __shared__ float    sB1v[HID];

    const int tid = threadIdx.x;

    // ---- TDM: DMA W3 into the (currently dead) sP region, wave 0 only ----
    if (tid < 32) {
        const unsigned lds_off = (unsigned)(uintptr_t)(void*)&sP[0]; // addr[31:0] == LDS offset
        tdm_load_w3(W3, lds_off);
        __builtin_amdgcn_s_wait_tensorcnt(0);
    }
    __syncthreads();   // W3 staged in LDS for all waves

    // ---- stage weights into WMMA B-fragment lane layout (f16) ----
// B 32x16 f16 layout: lane L -> n = L%16, kgroup = L/16;
// the 16 halves per lane are K = kgroup*8 + p (p<8), 16 + kgroup*8 + (p-8) (p>=8).
    {
        const float* w3s = (const float*)sP;    // staged raw W3 [k][col], stride 255
        for (int i = tid; i < NT3 * 32 * 16; i += BLOCK) {
            int nt   = i >> 9;            // /512
            int lane = (i >> 4) & 31;
            int p    = i & 15;
            int k    = ((p >> 3) << 4) + ((lane >> 4) << 3) + (p & 7);
            int col  = nt * 16 + (lane & 15);
            float v  = (col < PDIM) ? w3s[k * PDIM + col] : 0.f;
            sW3[i]   = (_Float16)v;
        }
    }
    for (int i = tid; i < 2 * 32 * 16; i += BLOCK) {
        int nt   = i >> 9;
        int lane = (i >> 4) & 31;
        int p    = i & 15;
        int k    = ((p >> 3) << 4) + ((lane >> 4) << 3) + (p & 7);
        int col  = nt * 16 + (lane & 15);
        sW2[i]   = (_Float16)W2[k * HID + col];
    }
    if (tid < 256) sB3[tid] = (tid < PDIM) ? b3[tid] : 0.f;
    if (tid < HID) { sB2[tid] = b2[tid]; sW1v[tid] = W1[tid]; sB1v[tid] = b1[tid]; }
    __syncthreads();   // also orders staging reads before params writes below

    const int lane = tid & 31;
    const int wave = tid >> 5;
    const int ln   = lane & 15;     // N-col within tile / M-row within tile
    const int lh   = lane >> 4;     // K-half (A,B) and row-half (C)
    const int eLoc = wave * 32 + lane;
    const int g    = blockIdx.x * ELEMS + eLoc;

    // ---- layer 1 (per-lane, 1->32) ----
    {
        float c0 = condx[g];
        #pragma unroll
        for (int j = 0; j < HID; ++j) {
            float hv = fmaxf(fmaf(c0, sW1v[j], sB1v[j]), 0.f);
            sH[eLoc * HSTR + j] = (_Float16)hv;
        }
    }
    // wave-private LDS region: same-wave DS ops are in order, no barrier needed.

    // ---- layers 2+3 via WMMA, two 16-element M-tiles per wave ----
    for (int mt = 0; mt < 2; ++mt) {
        const int mbase = wave * 32 + mt * 16;
        const _Float16* hrow = &sH[(mbase + ln) * HSTR];

        HFrag a;   // A = h1 tile (16x32 f16): lane row = ln, K chunks {8*lh.., 16+8*lh..}
        __builtin_memcpy(&a.q[0], hrow + 8 * lh,      16);
        __builtin_memcpy(&a.q[1], hrow + 16 + 8 * lh, 16);

        // layer 2: two 16x16 output tiles
        v8f acc2[2];
        #pragma unroll
        for (int nt = 0; nt < 2; ++nt) {
            HFrag b;
            const _Float16* bp = &sW2[(nt * 32 + lane) * 16];
            __builtin_memcpy(&b.q[0], bp,     16);
            __builtin_memcpy(&b.q[1], bp + 8, 16);
            v8f cz = {};
            acc2[nt] = __builtin_amdgcn_wmma_f32_16x16x32_f16(
                false, a.v, false, b.v, (short)0, cz, false, false);
        }
        // relu(h2 + b2) -> back into the same rows of sH (f16)
        #pragma unroll
        for (int nt = 0; nt < 2; ++nt) {
            const int   col  = nt * 16 + ln;
            const float bias = sB2[col];
            const int   row0 = mbase + 8 * lh;
            #pragma unroll
            for (int r = 0; r < 8; ++r) {
                float hv = fmaxf(acc2[nt][r] + bias, 0.f);
                sH[(row0 + r) * HSTR + col] = (_Float16)hv;
            }
        }

        HFrag a2;  // A = h2 tile
        __builtin_memcpy(&a2.q[0], hrow + 8 * lh,      16);
        __builtin_memcpy(&a2.q[1], hrow + 16 + 8 * lh, 16);

        // layer 3: 16 N-tiles, bias pre-splatted into C, result -> params^T in LDS
        const int erow0 = mbase + 8 * lh;   // element rows this lane's C covers
        #pragma unroll 4
        for (int nt = 0; nt < NT3; ++nt) {
            HFrag b;
            const _Float16* bp = &sW3[(nt * 32 + lane) * 16];
            __builtin_memcpy(&b.q[0], bp,     16);
            __builtin_memcpy(&b.q[1], bp + 8, 16);
            const int   col  = nt * 16 + ln;
            const float bias = sB3[col];
            v8f cc = {bias, bias, bias, bias, bias, bias, bias, bias};
            v8f d = __builtin_amdgcn_wmma_f32_16x16x32_f16(
                false, a2.v, false, b.v, (short)0, cc, false, false);
            float* pc = &sP[col * PSTR + erow0];
            f32x4 lo = {d[0], d[1], d[2], d[3]};
            f32x4 hi = {d[4], d[5], d[6], d[7]};
            __builtin_memcpy(pc,     &lo, 16);
            __builtin_memcpy(pc + 4, &hi, 16);
        }
    }

    // ---- spline (per lane, f32; params^T read column-parallel) ----
    #define P(kk) sP[(kk) * PSTR + eLoc]
    const float xv = x[g];
    const float xc = fminf(fmaxf(xv, -BOUND), BOUND);
    const bool  inside = (xv >= -BOUND) && (xv <= BOUND);

    float mw = -1e30f, mh = -1e30f;
    #pragma unroll 8
    for (int k = 0; k < K_BINS; ++k) {
        mw = fmaxf(mw, P(k));
        mh = fmaxf(mh, P(K_BINS + k));
    }
    float sw = 0.f, sh = 0.f;
    #pragma unroll 4
    for (int k = 0; k < K_BINS; ++k) {       // exp + cache back into LDS (lane owns col)
        float ew = __expf(P(k) - mw);          P(k) = ew;          sw += ew;
        float eh = __expf(P(K_BINS + k) - mh); P(K_BINS + k) = eh; sh += eh;
    }
    const float sclW = (1.f - MINW * K_BINS) * (2.f * BOUND) / sw;
    const float sclH = (1.f - MINH * K_BINS) * (2.f * BOUND) / sh;

    // bin search on cumulative widths (left edges)
    int   idx = 0;
    float xk = -BOUND, wk = 0.f, cum = -BOUND;
    #pragma unroll 4
    for (int k = 0; k < K_BINS; ++k) {
        float step = 2.f * BOUND * MINW + sclW * P(k);
        if (xc >= cum) { idx = k; xk = cum; wk = step; }
        cum += step;
    }
    // cumulative heights at idx
    float yk = -BOUND, hk = 0.f, ch = -BOUND;
    #pragma unroll 4
    for (int k = 0; k < K_BINS; ++k) {
        float step = 2.f * BOUND * MINH + sclH * P(K_BINS + k);
        if (k == idx) { yk = ch; hk = step; }
        ch += step;
    }

    // derivatives & lambda only at the selected bin
    float d0 = 1.f, d1 = 1.f;
    if (idx > 0)          d0 = MIND + softplus_f(P(2 * K_BINS + idx - 1));
    if (idx < K_BINS - 1) d1 = MIND + softplus_f(P(2 * K_BINS + idx));
    float lu  = P(3 * K_BINS - 1 + idx);
    float lam = 0.95f * (1.f / (1.f + __expf(-lu))) + 0.025f;

    float slope = hk / wk;
    float wb    = sqrtf(d0 / d1);
    float wcv   = (lam * d0 + (1.f - lam) * wb * d1) / slope;
    float ya    = yk, ybv = yk + hk;
    float yc    = ((1.f - lam) * ya + lam * wb * ybv) / ((1.f - lam) + lam * wb);
    float theta = (xc - xk) / wk;
    bool  left  = (theta <= lam);

    float num  = left ? (ya * (lam - theta) + wcv * yc * theta)
                      : (wcv * yc * (1.f - theta) + wb * ybv * (theta - lam));
    float den  = left ? ((lam - theta) + wcv * theta)
                      : (wcv * (1.f - theta) + wb * (theta - lam));
    float dnum = left ? (wcv * lam * (yc - ya))
                      : (wb * wcv * (1.f - lam) * (ybv - yc));

    float y_in  = num / den;
    float ld_in = __logf(dnum) - 2.f * __logf(fabsf(den)) - __logf(wk);

    out[g]     = inside ? y_in  : xv;
    out[n + g] = inside ? ld_in : 0.f;
    #undef P
}

extern "C" void kernel_launch(void* const* d_in, const int* in_sizes, int n_in,
                              void* d_out, int out_size, void* d_ws, size_t ws_size,
                              hipStream_t stream) {
    const float* xp = (const float*)d_in[0];
    const float* cx = (const float*)d_in[1];
    const float* W1 = (const float*)d_in[2];
    const float* b1 = (const float*)d_in[3];
    const float* W2 = (const float*)d_in[4];
    const float* b2 = (const float*)d_in[5];
    const float* W3 = (const float*)d_in[6];
    const float* b3 = (const float*)d_in[7];
    float* out = (float*)d_out;
    const int n = in_sizes[0];
    const int blocks = (n + ELEMS - 1) / ELEMS;
    hipLaunchKernelGGL(cond_spline_kernel, dim3(blocks), dim3(BLOCK), 0, stream,
                       xp, cx, W1, b1, W2, b2, W3, b3, out, n);
}
